// FlowmatchingActionHead_48120813584738
// MI455X (gfx1250) — compile-verified, matrix operand order checked by reading, not run on hardware
//
#include <hip/hip_runtime.h>
#include <hip/hip_bf16.h>

namespace {

constexpr int kInputDim  = 1536;
constexpr int kHiddenDim = 1024;
constexpr int kOutputDim = 1024;
constexpr int kBatch     = 128;
constexpr int kSeq       = 32;
constexpr int kCats      = 32;

typedef __attribute__((ext_vector_type(16))) __bf16          v16bf;
typedef __attribute__((ext_vector_type(8)))  __bf16          v8bf;
typedef __attribute__((ext_vector_type(8)))  float           v8f;
typedef __attribute__((ext_vector_type(16))) unsigned short  v16us;

__device__ __forceinline__ float bfbits_to_f32(unsigned short s) {
  unsigned int u = ((unsigned int)s) << 16;
  return __builtin_bit_cast(float, u);
}
__device__ __forceinline__ unsigned short f32_to_bfbits(float f) {
  unsigned int u = __builtin_bit_cast(unsigned int, f);
  u += 0x7FFFu + ((u >> 16) & 1u);   // round-to-nearest-even
  return (unsigned short)(u >> 16);
}
// fp32 -> bf16 hi + bf16 residual (lo). hi+lo carries ~16 mantissa bits.
__device__ __forceinline__ void split_bf16(float f, unsigned short& h, unsigned short& l) {
  h = f32_to_bfbits(f);
  l = f32_to_bfbits(f - bfbits_to_f32(h));
}

__device__ __forceinline__ v16bf concat8(v8bf a, v8bf b) {
  return __builtin_shufflevector(a, b, 0,1,2,3,4,5,6,7,8,9,10,11,12,13,14,15);
}

// ---------------------------------------------------------------------------
// pack_a: fp32 tensor -> two row-major bf16 planes (hi, lo). Pure streaming.
// ---------------------------------------------------------------------------
__global__ __launch_bounds__(256)
void pack_a_planes(const float* __restrict__ src,
                   unsigned short* __restrict__ hi,
                   unsigned short* __restrict__ lo,
                   int n4)   // number of float4 groups
{
  int i = blockIdx.x * 256 + threadIdx.x;
  if (i >= n4) return;
  float4 f = ((const float4*)src)[i];
  ushort4 h, l;
  split_bf16(f.x, h.x, l.x);
  split_bf16(f.y, h.y, l.y);
  split_bf16(f.z, h.z, l.z);
  split_bf16(f.w, h.w, l.w);
  ((ushort4*)hi)[i] = h;
  ((ushort4*)lo)[i] = l;
}

// ---------------------------------------------------------------------------
// pack_w: W (C, K, N) fp32 row-major  ->  WMMA B-fragment layout, bf16 hi/lo.
// Fragment (cat, nTile, kTile): 32 lanes x 16 bf16, contiguous (1 KB/plane).
// Lane l owns column n = nTile*16 + (l&15); kh = l>>4 selects the ISA K-half
// pattern: elems 0..7 -> K = kh*8+i, elems 8..15 -> K = 16+kh*8+(i-8).
// ---------------------------------------------------------------------------
__global__ __launch_bounds__(256)
void pack_w_frags(const float* __restrict__ W,
                  unsigned short* __restrict__ hi,
                  unsigned short* __restrict__ lo,
                  int K, int N)
{
  const int c     = blockIdx.y;
  const int nt    = blockIdx.x;          // 16-wide n tile
  const int numNT = N >> 4;
  const int numKT = K >> 5;
  const int lane  = threadIdx.x & 31;
  const int ktSub = threadIdx.x >> 5;    // 8 k-tiles per pass

  const int n  = nt * 16 + (lane & 15);
  const int kh = lane >> 4;
  const float* Wc = W + (size_t)c * (size_t)K * (size_t)N;

  for (int kt = ktSub; kt < numKT; kt += 8) {
    v16us hv, lv;
#pragma unroll
    for (int i = 0; i < 16; ++i) {
      const int k = kt * 32 + kh * 8 + (i < 8 ? i : i + 8);
      const float f = Wc[(size_t)k * N + n];
      unsigned short h, l;
      split_bf16(f, h, l);
      hv[i] = h;
      lv[i] = l;
    }
    const size_t fbase =
        (((size_t)c * numNT + nt) * (size_t)numKT + kt) * 512 + (size_t)lane * 16;
    *(v16us*)(hi + fbase) = hv;   // 32B contiguous per lane, fully coalesced
    *(v16us*)(lo + fbase) = lv;
  }
}

// ---------------------------------------------------------------------------
// Hot GEMM: D = A x W (+bias, opt ReLU) with pre-packed bf16 hi/lo operands.
// Block = 256 threads = 8 wave32s; block tile 32(M) x 64(N); K-step 32.
// Inner loop: 8x global_load_b128 + prefetch + 3x v_wmma_f32_16x16x32_bf16.
// No LDS, no barriers, no conversions. EXEC all-1s throughout.
// ---------------------------------------------------------------------------
template <bool RELU, bool PACK_OUT>
__global__ __launch_bounds__(256)
void cat_mlp_gemm_bf16x3(const unsigned short* __restrict__ Ahi,  // (B*32, K) bf16
                         const unsigned short* __restrict__ Alo,
                         const unsigned short* __restrict__ Wfh,  // fragment layout
                         const unsigned short* __restrict__ Wfl,
                         const float* __restrict__ bias,          // (C, N)
                         const int*   __restrict__ cat_ids,
                         float* __restrict__ outF,                // (B*32, N) if !PACK_OUT
                         unsigned short* __restrict__ outHi,      // planes if PACK_OUT
                         unsigned short* __restrict__ outLo,
                         int K, int N)
{
  const int batch = blockIdx.y;
  const int cat   = cat_ids[batch];
  const int tid   = threadIdx.x;
  const int lane  = tid & 31;
  const int wave  = tid >> 5;
  const int mt    = wave & 1;       // 16-row tile of M=32
  const int nt    = wave >> 1;      // 16-col tile within block
  const int col   = lane & 15;
  const int kh    = lane >> 4;
  const int numKT = K >> 5;
  const int numNT = N >> 4;
  const int ntG   = blockIdx.x * 4 + nt;

  const int aRow = mt * 16 + col;
  const unsigned short* aH = Ahi + (size_t)(batch * kSeq + aRow) * K + kh * 8;
  const unsigned short* aL = Alo + (size_t)(batch * kSeq + aRow) * K + kh * 8;

  const size_t fbase =
      (((size_t)cat * numNT + ntG) * (size_t)numKT) * 512 + (size_t)lane * 16;
  const unsigned short* bH = Wfh + fbase;
  const unsigned short* bL = Wfl + fbase;

  v8f acc = {0.f, 0.f, 0.f, 0.f, 0.f, 0.f, 0.f, 0.f};

#pragma unroll 2
  for (int kt = 0; kt < numKT; ++kt) {
    const unsigned short* ah0 = aH + kt * 32;
    const unsigned short* al0 = aL + kt * 32;
    v16bf ah = concat8(*(const v8bf*)(ah0), *(const v8bf*)(ah0 + 16));
    v16bf al = concat8(*(const v8bf*)(al0), *(const v8bf*)(al0 + 16));
    v16bf bh = *(const v16bf*)(bH + (size_t)kt * 512);
    v16bf bl = *(const v16bf*)(bL + (size_t)kt * 512);
    if (kt + 2 < numKT) {     // uniform branch; emits global_prefetch_b8
      __builtin_prefetch((const void*)(bH + (size_t)(kt + 2) * 512), 0, 1);
    }
    // 3-term bf16 emulation of fp32 GEMM (drops only lo*lo, ~2^-16 rel)
    acc = __builtin_amdgcn_wmma_f32_16x16x32_bf16(false, ah, false, bh, (short)0, acc, false, false);
    acc = __builtin_amdgcn_wmma_f32_16x16x32_bf16(false, al, false, bh, (short)0, acc, false, false);
    acc = __builtin_amdgcn_wmma_f32_16x16x32_bf16(false, ah, false, bl, (short)0, acc, false, false);
  }

  // Epilogue per 16x16 f32 C/D layout: lane = column, VGPR r = row mBase+r.
  const int nCol = ntG * 16 + col;
  const float bv = bias[(size_t)cat * N + nCol];
  const int mBase = mt * 16 + kh * 8;
#pragma unroll
  for (int r = 0; r < 8; ++r) {
    float v = acc[r] + bv;
    if (RELU) v = fmaxf(v, 0.f);
    const size_t off = (size_t)(batch * kSeq + mBase + r) * N + nCol;
    if (PACK_OUT) {
      unsigned short h, l;
      split_bf16(v, h, l);
      outHi[off] = h;
      outLo[off] = l;
    } else {
      outF[off] = v;
    }
  }
}

// ---------------------------------------------------------------------------
// Fallback (small ws): fused per-tile split, LDS-staged W (round-1 kernel).
// ---------------------------------------------------------------------------
template <bool RELU>
__global__ __launch_bounds__(256)
void cat_mlp_gemm_fused(const float* __restrict__ A,
                        const float* __restrict__ W,
                        const float* __restrict__ bias,
                        const int*   __restrict__ cat_ids,
                        float* __restrict__ out,
                        int K, int N)
{
  constexpr int KT  = 32;
  constexpr int NT  = 64;
  constexpr int LDB = 68;
  __shared__ float sB[KT * LDB];

  const int batch = blockIdx.y;
  const int n0    = blockIdx.x * NT;
  const int cat   = cat_ids[batch];
  const int tid   = threadIdx.x;
  const int lane  = tid & 31;
  const int wave  = tid >> 5;
  const int mt    = wave & 1;
  const int nt    = wave >> 1;
  const int col   = lane & 15;
  const int kh    = lane >> 4;

  const float* Ab = A + (size_t)batch * kSeq * K;
  const float* Wc = W + (size_t)cat * (size_t)K * (size_t)N + n0;
  const float* Arow = Ab + (size_t)(mt * 16 + col) * K + kh * 8;

  const int kr = tid >> 3;
  const int nc = (tid & 7) * 8;

  v8f acc = {0.f, 0.f, 0.f, 0.f, 0.f, 0.f, 0.f, 0.f};

  for (int k0 = 0; k0 < K; k0 += KT) {
    {
      const float* src = Wc + (size_t)(k0 + kr) * N + nc;
      float4 p0 = *(const float4*)(src);
      float4 p1 = *(const float4*)(src + 4);
      *(float4*)(&sB[kr * LDB + nc])     = p0;
      *(float4*)(&sB[kr * LDB + nc + 4]) = p1;
    }
    __syncthreads();

    const float* ap = Arow + k0;
    float4 a0 = *(const float4*)(ap);
    float4 a1 = *(const float4*)(ap + 4);
    float4 a2 = *(const float4*)(ap + 16);
    float4 a3 = *(const float4*)(ap + 20);
    float af[16] = {a0.x,a0.y,a0.z,a0.w, a1.x,a1.y,a1.z,a1.w,
                    a2.x,a2.y,a2.z,a2.w, a3.x,a3.y,a3.z,a3.w};

    float bfr[16];
    const int bcol = nt * 16 + col;
#pragma unroll
    for (int i = 0; i < 8; ++i) {
      bfr[i]     = sB[(kh * 8 + i) * LDB + bcol];
      bfr[i + 8] = sB[(16 + kh * 8 + i) * LDB + bcol];
    }

    v16bf ah, al, bh, bl;
#pragma unroll
    for (int i = 0; i < 16; ++i) {
      unsigned short h, l, g, m;
      split_bf16(af[i], h, l);
      split_bf16(bfr[i], g, m);
      ah[i] = __builtin_bit_cast(__bf16, h);
      al[i] = __builtin_bit_cast(__bf16, l);
      bh[i] = __builtin_bit_cast(__bf16, g);
      bl[i] = __builtin_bit_cast(__bf16, m);
    }

    acc = __builtin_amdgcn_wmma_f32_16x16x32_bf16(false, ah, false, bh, (short)0, acc, false, false);
    acc = __builtin_amdgcn_wmma_f32_16x16x32_bf16(false, al, false, bh, (short)0, acc, false, false);
    acc = __builtin_amdgcn_wmma_f32_16x16x32_bf16(false, ah, false, bl, (short)0, acc, false, false);

    __syncthreads();
  }

  const int nCol = n0 + nt * 16 + col;
  const float bv = bias[(size_t)cat * N + nCol];
  float* op = out + (size_t)batch * kSeq * N + nCol;
  const int mBase = mt * 16 + kh * 8;
#pragma unroll
  for (int r = 0; r < 8; ++r) {
    float v = acc[r] + bv;
    if (RELU) v = fmaxf(v, 0.f);
    op[(size_t)(mBase + r) * N] = v;
  }
}

// ws layout (bytes) for the packed path
constexpr size_t kSzW1Plane = (size_t)kCats * kInputDim  * kHiddenDim * 2;  // 100663296
constexpr size_t kSzW2Plane = (size_t)kCats * kHiddenDim * kOutputDim * 2;  //  67108864
constexpr size_t kSzXPlane  = (size_t)kBatch * kSeq * kInputDim  * 2;       //  12582912
constexpr size_t kSzHPlane  = (size_t)kBatch * kSeq * kHiddenDim * 2;       //   8388608
constexpr size_t kWsNeeded  = 2 * (kSzW1Plane + kSzW2Plane + kSzXPlane + kSzHPlane);

} // namespace

extern "C" void kernel_launch(void* const* d_in, const int* in_sizes, int n_in,
                              void* d_out, int out_size, void* d_ws, size_t ws_size,
                              hipStream_t stream) {
  // setup_inputs order: x, cat_ids, W1, b1, W2, b2
  const float* x   = (const float*)d_in[0];
  const int*   cid = (const int*)  d_in[1];
  const float* W1  = (const float*)d_in[2];
  const float* b1  = (const float*)d_in[3];
  const float* W2  = (const float*)d_in[4];
  const float* b2  = (const float*)d_in[5];
  float* out = (float*)d_out;

  dim3 block(256, 1, 1);

  if (ws_size >= kWsNeeded) {
    char* p = (char*)d_ws;
    unsigned short* W1h = (unsigned short*)p;                 p += kSzW1Plane;
    unsigned short* W1l = (unsigned short*)p;                 p += kSzW1Plane;
    unsigned short* W2h = (unsigned short*)p;                 p += kSzW2Plane;
    unsigned short* W2l = (unsigned short*)p;                 p += kSzW2Plane;
    unsigned short* Xh  = (unsigned short*)p;                 p += kSzXPlane;
    unsigned short* Xl  = (unsigned short*)p;                 p += kSzXPlane;
    unsigned short* Hh  = (unsigned short*)p;                 p += kSzHPlane;
    unsigned short* Hl  = (unsigned short*)p;

    const int n4x = kBatch * kSeq * kInputDim / 4;
    pack_a_planes<<<dim3((n4x + 255) / 256), block, 0, stream>>>(x, Xh, Xl, n4x);
    pack_w_frags<<<dim3(kHiddenDim / 16, kCats), block, 0, stream>>>(W1, W1h, W1l, kInputDim, kHiddenDim);
    pack_w_frags<<<dim3(kOutputDim / 16, kCats), block, 0, stream>>>(W2, W2h, W2l, kHiddenDim, kOutputDim);

    cat_mlp_gemm_bf16x3<true, true><<<dim3(kHiddenDim / 64, kBatch), block, 0, stream>>>(
        Xh, Xl, W1h, W1l, b1, cid, nullptr, Hh, Hl, kInputDim, kHiddenDim);
    cat_mlp_gemm_bf16x3<false, false><<<dim3(kOutputDim / 64, kBatch), block, 0, stream>>>(
        Hh, Hl, W2h, W2l, b2, cid, out, nullptr, nullptr, kHiddenDim, kOutputDim);
  } else {
    float* hidden = (float*)d_ws;  // 16.78 MB fp32 intermediate
    cat_mlp_gemm_fused<true ><<<dim3(kHiddenDim / 64, kBatch), block, 0, stream>>>(
        x, W1, b1, cid, hidden, kInputDim, kHiddenDim);
    cat_mlp_gemm_fused<false><<<dim3(kOutputDim / 64, kBatch), block, 0, stream>>>(
        hidden, W2, b2, cid, out, kHiddenDim, kOutputDim);
  }

  (void)in_sizes; (void)n_in; (void)out_size;
}
